// LineCNNTransformer_43439299231974
// MI455X (gfx1250) — compile-verified, matrix operand order checked by reading, not run on hardware
//
#include <hip/hip_runtime.h>
#include <hip/hip_bf16.h>
#include <math.h>

typedef __attribute__((ext_vector_type(16))) _Float16 v16h;
typedef __attribute__((ext_vector_type(8)))  float    v8f;

#define Dm   256
#define Hh   4
#define HDm  64
#define NL   4
#define FFDm 256
#define Sm   128
#define Cm   90
#define Bm   64
#define SXm  512
#define T_START 1
#define T_END   2
#define T_PAD   3
#define NEGF   (-1e9f)
#define SQRTD  16.0f

union Frag { v16h v; unsigned u[8]; };
union H2   { _Float16 h[2]; unsigned u; };

// sinusoidal positional encoding value, matches reference _pos_encoding
__device__ __forceinline__ float pe_val(int pos, int d) {
    int i2 = d & ~1;
    float freq = __expf(-(float)i2 * (9.21034037198e0f / 256.0f)); // ln(10000)/D
    float a = (float)pos * freq;
    return (d & 1) ? __cosf(a) : __sinf(a);
}

// ---- CDNA5 async copy: contiguous global tile -> LDS, 16B/lane chunks (ASYNCcnt) ----
// 128-thread block; nbytes must be a multiple of 2048.
__device__ __forceinline__ void async_tile_copy(const void* gbase, unsigned goff0,
                                                void* ldsdst, int t, int nbytes) {
    unsigned lbase = (unsigned)(size_t)ldsdst;   // low 32 bits of LDS aperture addr = LDS offset
    for (int i = 0; i < nbytes; i += 2048) {
        unsigned off = (unsigned)i + (unsigned)t * 16u;
        asm volatile("global_load_async_to_lds_b128 %0, %1, %2"
                     :: "v"(lbase + off), "v"(goff0 + off), "s"(gbase)
                     : "memory");
    }
    asm volatile("s_wait_asynccnt 0x0" ::: "memory");
}

// ---------------- weight conversion f32 -> f16 ----------------
__global__ void k_cvt_f16(const float* __restrict__ src, _Float16* __restrict__ dst, int n) {
    int i = blockIdx.x * blockDim.x + threadIdx.x;
    if (i < n) dst[i] = (_Float16)src[i];
}

// ---------------- build memory (SX, B, D) as f16 row-major -----------
__global__ void k_build_mem(const float* __restrict__ mf, _Float16* __restrict__ mem16) {
    int d = blockIdx.x;
    int b = blockIdx.y;
    for (int s = threadIdx.x; s < SXm; s += blockDim.x) {
        float v = mf[((size_t)b * Dm + d) * SXm + s] * SQRTD + pe_val(s, d);
        mem16[((size_t)s * Bm + b) * Dm + d] = (_Float16)v;
    }
}

// ---------------- init tokens / key mask ----------------
__global__ void k_init(int* __restrict__ tok, float* __restrict__ keymask) {
    int i = blockIdx.x * blockDim.x + threadIdx.x;
    if (i < Bm * Sm) {
        int s = i % Sm;
        tok[i] = (s == 0) ? T_START : T_PAD;
        keymask[i] = (s == 0) ? 0.0f : NEGF;
    }
}

// ---------------- token embedding for position p (used once, p=0) ----------------
__global__ void k_embed(const int* __restrict__ tok, const float* __restrict__ emb,
                        float* __restrict__ x, int p) {
    int b = blockIdx.x, d = threadIdx.x;
    int t = tok[b * Sm + p];
    x[b * Dm + d] = emb[t * Dm + d] * SQRTD + pe_val(p, d);
}

// ================= WMMA GEMM (f16 A in global): out = A[M][256] * W[N][256]^T ============
// Block = 128 threads = 4 waves; tile = 16(M) x 64(N); A staged via async-to-LDS.
// mode bit0: exact GELU -> outH; bit1: out = resid + acc -> outF.
__global__ void k_gemm(const _Float16* __restrict__ A, const _Float16* __restrict__ W,
                       const float* __restrict__ bias, const float* __restrict__ resid,
                       float* __restrict__ outF, _Float16* __restrict__ outH,
                       int N, int mode) {
    __shared__ unsigned lA[16 * 128];          // 16 rows x 256 halves = 8 KB
    const unsigned* Wu = (const unsigned*)W;
    int m0 = blockIdx.y * 16;
    int t = threadIdx.x;
    async_tile_copy(A, (unsigned)m0 * 512u, lA, t, 8192);   // 16 contiguous rows
    __syncthreads();

    int wave = t >> 5, lane = t & 31;
    int n0 = blockIdx.x * 64 + wave * 16;
    int nr = n0 + (lane & 15);
    int hi = (lane >= 16) ? 8 : 0;
    int mrow = lane & 15;
    const unsigned* wrow = Wu + (size_t)nr * 128;

    Frag fa, fb;
    v8f acc = {0.f, 0.f, 0.f, 0.f, 0.f, 0.f, 0.f, 0.f};
    #pragma unroll
    for (int kb = 0; kb < 8; kb++) {
        int k0 = kb * 32;
        __builtin_prefetch(wrow + ((k0 + 32) & 255) / 2, 0, 1);
        #pragma unroll
        for (int v = 0; v < 8; v++) {
            int kk = ((v < 4) ? 2 * v : 16 + 2 * (v - 4)) + hi;
            int ki = (k0 + kk) >> 1;
            fa.u[v] = lA[mrow * 128 + ki];
            fb.u[v] = wrow[ki];
        }
        acc = __builtin_amdgcn_wmma_f32_16x16x32_f16(false, fa.v, false, fb.v,
                                                     (short)0, acc, false, false);
    }
    float bv = bias ? bias[nr] : 0.0f;
    #pragma unroll
    for (int r = 0; r < 8; r++) {
        int row = m0 + r + hi;
        size_t o = (size_t)row * N + nr;
        float val = acc[r] + bv;
        if (mode & 1) val = 0.5f * val * (1.0f + erff(val * 0.70710678118654752f));
        if (mode & 2) val += resid[o];
        if (outF) outF[o] = val;
        if (outH) outH[o] = (_Float16)val;
    }
}

// ========= fused LayerNorm + WMMA GEMM (f32 X in global) =========
// mode 0: outF = acc+bias ; mode 1: outH = gelu(acc+bias) ;
// mode 4: QKV split -> q to qf (f32), k/v appended to caches at position p (f16).
__global__ void k_lngemm(const float* __restrict__ X, const float* __restrict__ g,
                         const float* __restrict__ bb,
                         const _Float16* __restrict__ W, const float* __restrict__ bias,
                         float* __restrict__ outF, _Float16* __restrict__ outH,
                         float* __restrict__ qf, _Float16* __restrict__ kc,
                         _Float16* __restrict__ vc, int p,
                         int N, int mode) {
    __shared__ float    xs[16 * 256];          // 16 KB raw f32 rows
    __shared__ unsigned lA[16 * 128];          // 8 KB normalized f16 tile
    __shared__ float    rred[128];
    __shared__ float    rmean[16], rrstd[16];
    int m0 = blockIdx.y * 16;
    int t = threadIdx.x;
    async_tile_copy(X, (unsigned)m0 * 1024u, xs, t, 16384);  // 16 contiguous f32 rows
    __syncthreads();

    // per-row LayerNorm stats: 8 threads per row, 32 elements each
    int r = t >> 3, c0 = (t & 7) * 32;
    float s = 0.f;
    for (int c = 0; c < 32; c++) s += xs[r * 256 + c0 + c];
    rred[t] = s; __syncthreads();
    if ((t & 7) == 0) {
        float m = 0.f;
        for (int i = 0; i < 8; i++) m += rred[t + i];
        rmean[r] = m * (1.0f / 256.0f);
    }
    __syncthreads();
    float mean = rmean[r];
    s = 0.f;
    for (int c = 0; c < 32; c++) { float d = xs[r * 256 + c0 + c] - mean; s += d * d; }
    rred[t] = s; __syncthreads();
    if ((t & 7) == 0) {
        float vv = 0.f;
        for (int i = 0; i < 8; i++) vv += rred[t + i];
        rrstd[r] = rsqrtf(vv * (1.0f / 256.0f) + 1e-5f);
    }
    __syncthreads();
    float rstd = rrstd[r];
    for (int c = c0; c < c0 + 32; c += 2) {
        H2 h2;
        h2.h[0] = (_Float16)((xs[r * 256 + c]     - mean) * rstd * g[c]     + bb[c]);
        h2.h[1] = (_Float16)((xs[r * 256 + c + 1] - mean) * rstd * g[c + 1] + bb[c + 1]);
        lA[r * 128 + (c >> 1)] = h2.u;
    }
    __syncthreads();

    int wave = t >> 5, lane = t & 31;
    int n0 = blockIdx.x * 64 + wave * 16;
    int nr = n0 + (lane & 15);
    int hi = (lane >= 16) ? 8 : 0;
    int mrow = lane & 15;
    const unsigned* Wu = (const unsigned*)W;
    const unsigned* wrow = Wu + (size_t)nr * 128;

    Frag fa, fb;
    v8f acc = {0.f, 0.f, 0.f, 0.f, 0.f, 0.f, 0.f, 0.f};
    #pragma unroll
    for (int kb = 0; kb < 8; kb++) {
        int k0 = kb * 32;
        __builtin_prefetch(wrow + ((k0 + 32) & 255) / 2, 0, 1);
        #pragma unroll
        for (int v = 0; v < 8; v++) {
            int kk = ((v < 4) ? 2 * v : 16 + 2 * (v - 4)) + hi;
            int ki = (k0 + kk) >> 1;
            fa.u[v] = lA[mrow * 128 + ki];
            fb.u[v] = wrow[ki];
        }
        acc = __builtin_amdgcn_wmma_f32_16x16x32_f16(false, fa.v, false, fb.v,
                                                     (short)0, acc, false, false);
    }
    float bv = bias ? bias[nr] : 0.0f;
    #pragma unroll
    for (int r8 = 0; r8 < 8; r8++) {
        int row = m0 + r8 + hi;
        float val = acc[r8] + bv;
        if (mode == 4) {                       // QKV epilogue: q -> f32, k/v -> caches
            if (nr < 256)       qf[(size_t)row * 256 + nr] = val;
            else if (nr < 512)  kc[((size_t)row * Sm + p) * Dm + (nr - 256)] = (_Float16)val;
            else                vc[((size_t)row * Sm + p) * Dm + (nr - 512)] = (_Float16)val;
        } else {
            if (mode & 1) val = 0.5f * val * (1.0f + erff(val * 0.70710678118654752f));
            size_t o = (size_t)row * N + nr;
            if (outF) outF[o] = val;
            if (outH) outH[o] = (_Float16)val;
        }
    }
}

// ---------------- causal self-attention for query position p ----------------
__global__ void k_attn_self(const float* __restrict__ qf, const _Float16* __restrict__ kc,
                            const _Float16* __restrict__ vc, const float* __restrict__ keymask,
                            _Float16* __restrict__ outh, int p) {
    __shared__ float sc[Sm];
    __shared__ float qs[HDm];
    __shared__ float red[64];
    int b = blockIdx.x / Hh, h = blockIdx.x % Hh;
    int t = threadIdx.x;                          // 64 threads
    qs[t] = qf[b * Dm + h * HDm + t];
    __syncthreads();
    float lmax = -3.4e38f;
    for (int j = t; j <= p; j += 64) {
        const _Float16* kp = kc + ((size_t)b * Sm + j) * Dm + h * HDm;
        float s = 0.f;
        for (int d2 = 0; d2 < HDm; d2++) s += qs[d2] * (float)kp[d2];
        s = s * 0.125f + keymask[b * Sm + j];
        sc[j] = s; lmax = fmaxf(lmax, s);
    }
    red[t] = lmax; __syncthreads();
    for (int o = 32; o > 0; o >>= 1) { if (t < o) red[t] = fmaxf(red[t], red[t + o]); __syncthreads(); }
    float mx = red[0]; __syncthreads();
    float lsum = 0.f;
    for (int j = t; j <= p; j += 64) { float e = __expf(sc[j] - mx); sc[j] = e; lsum += e; }
    red[t] = lsum; __syncthreads();
    for (int o = 32; o > 0; o >>= 1) { if (t < o) red[t] += red[t + o]; __syncthreads(); }
    float inv = 1.0f / red[0]; __syncthreads();
    float acc = 0.f;
    for (int j = 0; j <= p; j++)
        acc += sc[j] * (float)vc[((size_t)b * Sm + j) * Dm + h * HDm + t];
    outh[b * Dm + h * HDm + t] = (_Float16)(acc * inv);
}

// ---------------- cross-attention over precomputed (K|V), kv row = (s*B+b)*512 ------
__global__ void k_attn_cross(const float* __restrict__ q, const _Float16* __restrict__ kv,
                             _Float16* __restrict__ outh) {
    __shared__ float sc[SXm];
    __shared__ float qs[HDm];
    __shared__ float red[256];
    int b = blockIdx.x / Hh, h = blockIdx.x % Hh;
    int t = threadIdx.x;                          // 256 threads
    if (t < HDm) qs[t] = q[b * Dm + h * HDm + t];
    __syncthreads();
    float lmax = -3.4e38f;
    for (int j = t; j < SXm; j += 256) {
        const _Float16* kp = kv + ((size_t)j * Bm + b) * 512 + h * HDm;
        float s = 0.f;
        for (int d2 = 0; d2 < HDm; d2++) s += qs[d2] * (float)kp[d2];
        s *= 0.125f;
        sc[j] = s; lmax = fmaxf(lmax, s);
    }
    red[t] = lmax; __syncthreads();
    for (int o = 128; o > 0; o >>= 1) { if (t < o) red[t] = fmaxf(red[t], red[t + o]); __syncthreads(); }
    float mx = red[0]; __syncthreads();
    float lsum = 0.f;
    for (int j = t; j < SXm; j += 256) { float e = __expf(sc[j] - mx); sc[j] = e; lsum += e; }
    red[t] = lsum; __syncthreads();
    for (int o = 128; o > 0; o >>= 1) { if (t < o) red[t] += red[t + o]; __syncthreads(); }
    float inv = 1.0f / red[0]; __syncthreads();
    int d = t & 63, part = t >> 6;
    float acc = 0.f;
    for (int j = part * 128; j < part * 128 + 128; j++)
        acc += sc[j] * (float)kv[((size_t)j * Bm + b) * 512 + 256 + h * HDm + d];
    red[t] = acc; __syncthreads();
    if (t < 64) {
        float tot = red[t] + red[t + 64] + red[t + 128] + red[t + 192];
        outh[b * Dm + h * HDm + t] = (_Float16)(tot * inv);
    }
}

// ------ final projection + argmax -> token[b][sy]; fused embedding of position sy ------
__global__ void k_logits_embed(const float* __restrict__ x, const float* __restrict__ fcw,
                               const float* __restrict__ fcb, const float* __restrict__ emb,
                               int* __restrict__ tok, float* __restrict__ keymask,
                               float* __restrict__ xnext, int sy) {
    __shared__ float xs[Dm];
    __shared__ float bv[128];
    __shared__ int   bi[128];
    __shared__ int   stok;
    int b = blockIdx.x, t = threadIdx.x;          // 128 threads
    xs[t] = x[b * Dm + t];
    xs[t + 128] = x[b * Dm + t + 128];
    __syncthreads();
    float val = -3.4e38f;
    if (t < Cm) {
        float s = fcb[t];
        for (int d2 = 0; d2 < Dm; d2++) s += xs[d2] * fcw[t * Dm + d2];
        val = s;
    }
    bv[t] = val; bi[t] = t; __syncthreads();
    for (int o = 64; o > 0; o >>= 1) {
        if (t < o) {
            if (bv[t + o] > bv[t] || (bv[t + o] == bv[t] && bi[t + o] < bi[t])) {
                bv[t] = bv[t + o]; bi[t] = bi[t + o];
            }
        }
        __syncthreads();
    }
    if (t == 0) {
        int tk = bi[0];
        stok = tk;
        tok[b * Sm + sy] = tk;
        keymask[b * Sm + sy] = (tk == T_PAD) ? NEGF : 0.0f;
    }
    __syncthreads();
    int tk = stok;                                // embed token at position sy for next step
    xnext[b * Dm + t]       = emb[tk * Dm + t]       * SQRTD + pe_val(sy, t);
    xnext[b * Dm + t + 128] = emb[tk * Dm + t + 128] * SQRTD + pe_val(sy, t + 128);
}

// ---------------- propagate PAD after first END/PAD ----------------
__global__ void k_finalize(int* __restrict__ tok) {
    int b = threadIdx.x;
    if (b < Bm) {
        int done = 0;
        for (int j = 0; j < Sm - 1; j++) {
            int v = tok[b * Sm + j];
            if (v == T_END || v == T_PAD) done = 1;
            if (done) tok[b * Sm + j + 1] = T_PAD;
        }
    }
}

extern "C" void kernel_launch(void* const* d_in, const int* in_sizes, int n_in,
                              void* d_out, int out_size, void* d_ws, size_t ws_size,
                              hipStream_t stream) {
    const float* memory_feat = (const float*)d_in[0];
    const float* emb   = (const float*)d_in[1];
    const float* fc_w  = (const float*)d_in[2];
    const float* fc_b  = (const float*)d_in[3];
    const float* sa_w  = (const float*)d_in[4];
    const float* sa_b  = (const float*)d_in[5];
    const float* sa_ow = (const float*)d_in[6];
    const float* sa_ob = (const float*)d_in[7];
    const float* ca_w  = (const float*)d_in[8];
    const float* ca_b  = (const float*)d_in[9];
    const float* ca_ow = (const float*)d_in[10];
    const float* ca_ob = (const float*)d_in[11];
    const float* f_w1  = (const float*)d_in[12];
    const float* f_b1  = (const float*)d_in[13];
    const float* f_w2  = (const float*)d_in[14];
    const float* f_b2  = (const float*)d_in[15];
    const float* ln_g  = (const float*)d_in[16];
    const float* ln_b  = (const float*)d_in[17];
    (void)in_sizes; (void)n_in; (void)out_size; (void)ws_size;

    size_t off = 0;
    char* base = (char*)d_ws;
    auto carve = [&](size_t bytes) -> char* {
        char* p = base + off;
        off = (off + bytes + 255) & ~(size_t)255;
        return p;
    };
    _Float16* sa_w16  = (_Float16*)carve((size_t)NL * 768 * 256 * 2);
    _Float16* sa_ow16 = (_Float16*)carve((size_t)NL * 256 * 256 * 2);
    _Float16* ca_w16  = (_Float16*)carve((size_t)NL * 768 * 256 * 2);
    _Float16* ca_ow16 = (_Float16*)carve((size_t)NL * 256 * 256 * 2);
    _Float16* fw1_16  = (_Float16*)carve((size_t)NL * 256 * 256 * 2);
    _Float16* fw2_16  = (_Float16*)carve((size_t)NL * 256 * 256 * 2);
    _Float16* mem16   = (_Float16*)carve((size_t)SXm * Bm * Dm * 2);
    _Float16* kvc     = (_Float16*)carve((size_t)NL * SXm * Bm * 512 * 2);  // 134 MB, L2-resident
    _Float16* kself   = (_Float16*)carve((size_t)NL * Bm * Sm * Dm * 2);
    _Float16* vself   = (_Float16*)carve((size_t)NL * Bm * Sm * Dm * 2);
    float*    x       = (float*)carve((size_t)Bm * Dm * 4);
    float*    qf      = (float*)carve((size_t)Bm * Dm * 4);
    _Float16* attnh   = (_Float16*)carve((size_t)Bm * Dm * 2);
    float*    qc      = (float*)carve((size_t)Bm * Dm * 4);
    _Float16* h16     = (_Float16*)carve((size_t)Bm * FFDm * 2);
    float*    keymask = (float*)carve((size_t)Bm * Sm * 4);
    int*      tok     = (int*)d_out;

    auto cvt = [&](const float* s, _Float16* d, int n) {
        k_cvt_f16<<<(n + 255) / 256, 256, 0, stream>>>(s, d, n);
    };
    cvt(sa_w,  sa_w16,  NL * 768 * 256);
    cvt(sa_ow, sa_ow16, NL * 256 * 256);
    cvt(ca_w,  ca_w16,  NL * 768 * 256);
    cvt(ca_ow, ca_ow16, NL * 256 * 256);
    cvt(f_w1,  fw1_16,  NL * 256 * 256);
    cvt(f_w2,  fw2_16,  NL * 256 * 256);

    // encode(): memory as f16, then one-time cross K|V per layer (big WMMA GEMM)
    k_build_mem<<<dim3(Dm, Bm), 128, 0, stream>>>(memory_feat, mem16);
    for (int l = 0; l < NL; l++) {
        k_gemm<<<dim3(512 / 64, (SXm * Bm) / 16), 128, 0, stream>>>(
            mem16, ca_w16 + (size_t)l * 768 * 256 + 256 * 256,
            ca_b + l * 768 + 256, nullptr,
            nullptr, kvc + (size_t)l * SXm * Bm * 512, 512, 0);
    }

    k_init<<<(Bm * Sm + 255) / 256, 256, 0, stream>>>(tok, keymask);
    k_embed<<<Bm, 256, 0, stream>>>(tok, emb, x, 0);

    for (int sy = 1; sy < Sm; sy++) {
        int p = sy - 1;
        for (int l = 0; l < NL; l++) {
            _Float16* kcl = kself + (size_t)l * Bm * Sm * Dm;
            _Float16* vcl = vself + (size_t)l * Bm * Sm * Dm;
            // self-attention: fused LN1 + QKV (q->qf, k/v appended to caches)
            k_lngemm<<<dim3(768 / 64, Bm / 16), 128, 0, stream>>>(
                x, ln_g + (l * 3 + 0) * Dm, ln_b + (l * 3 + 0) * Dm,
                sa_w16 + (size_t)l * 768 * 256, sa_b + l * 768,
                nullptr, nullptr, qf, kcl, vcl, p, 768, 4);
            k_attn_self<<<Bm * Hh, 64, 0, stream>>>(qf, kcl, vcl, keymask, attnh, p);
            k_gemm<<<dim3(4, Bm / 16), 128, 0, stream>>>(
                attnh, sa_ow16 + (size_t)l * 256 * 256, sa_ob + l * 256,
                x, x, nullptr, 256, 2);
            // cross-attention: fused LN2 + Q proj; K,V precomputed
            k_lngemm<<<dim3(4, Bm / 16), 128, 0, stream>>>(
                x, ln_g + (l * 3 + 1) * Dm, ln_b + (l * 3 + 1) * Dm,
                ca_w16 + (size_t)l * 768 * 256, ca_b + l * 768,
                qc, nullptr, nullptr, nullptr, nullptr, 0, 256, 0);
            k_attn_cross<<<Bm * Hh, 256, 0, stream>>>(qc, kvc + (size_t)l * SXm * Bm * 512, attnh);
            k_gemm<<<dim3(4, Bm / 16), 128, 0, stream>>>(
                attnh, ca_ow16 + (size_t)l * 256 * 256, ca_ob + l * 256,
                x, x, nullptr, 256, 2);
            // feed-forward: fused LN3 + FF1(GELU), then FF2 + residual
            k_lngemm<<<dim3(4, Bm / 16), 128, 0, stream>>>(
                x, ln_g + (l * 3 + 2) * Dm, ln_b + (l * 3 + 2) * Dm,
                fw1_16 + (size_t)l * 256 * 256, f_b1 + l * 256,
                nullptr, h16, nullptr, nullptr, nullptr, 0, 256, 1);
            k_gemm<<<dim3(4, Bm / 16), 128, 0, stream>>>(
                h16, fw2_16 + (size_t)l * 256 * 256, f_b2 + l * 256,
                x, x, nullptr, 256, 2);
        }
        // logits + argmax + fused embedding of the new token (writes x for next step)
        k_logits_embed<<<Bm, 128, 0, stream>>>(x, fc_w, fc_b, emb, tok, keymask, x, sy);
    }
    k_finalize<<<1, 64, 0, stream>>>(tok);
}
